// ProteinGNNoGlobal_24438363914613
// MI455X (gfx1250) — matrix operand order repaired and checked by reading, not compiled, not run on hardware
//
#include <hip/hip_runtime.h>

// ---------------- problem constants (match reference) ----------------
#define GN_N   100000
#define GN_E   1200000
#define GN_E2  2400000   // duplicated (symmetrized) edges
#define GN_G   400

typedef __attribute__((ext_vector_type(16))) _Float16 v16h;
typedef __attribute__((ext_vector_type(8)))  _Float16 h8;
typedef __attribute__((ext_vector_type(8)))  float    v8f;
typedef __attribute__((ext_vector_type(8)))  float    f8;

__device__ __forceinline__ float gn_sigmoid(float x) { return 1.0f / (1.0f + expf(-x)); }

// ---------------------------------------------------------------------
// Weight packing in WMMA B-fragment order: out[t][q][lane][h] so each
// lane's 16-half B fragment is one contiguous 32-byte load.
//   k = q*32 + (h<8 ? h : h+8) + (lane>=16 ? 8 : 0), col = t*16 + (lane&15)
// rows = concat(W1[r1 x cols], W2[r2 x cols]), zero-padded to KP rows.
// ---------------------------------------------------------------------
__global__ void pack_weights_frag(const float* __restrict__ W1, int r1,
                                  const float* __restrict__ W2, int r2,
                                  int cols, int KP, _Float16* __restrict__ out) {
    int QK = KP / 32, NT = cols / 16;
    int total = NT * QK * 32 * 16;
    int i = blockIdx.x * blockDim.x + threadIdx.x;
    if (i >= total) return;
    int h    = i & 15;
    int lane = (i >> 4) & 31;
    int q    = (i >> 9) % QK;
    int t    = i / (QK * 512);
    int k = q * 32 + ((h < 8) ? h : h + 8) + ((lane >= 16) ? 8 : 0);
    int c = t * 16 + (lane & 15);
    float v = 0.0f;
    if (k < r1)            v = W1[k * cols + c];
    else if (k < r1 + r2)  v = W2[(k - r1) * cols + c];
    out[i] = (_Float16)v;
}

// ---------------------------------------------------------------------
// Edge encoder: rbf(16) ++ edge_raw(2) -> 4 -> 8 (ReLU), duplicated edges.
// ---------------------------------------------------------------------
__global__ void edge_encoder(const float* __restrict__ edge_raw,
                             const float* __restrict__ distances,
                             const float* __restrict__ We1, const float* __restrict__ be1,
                             const float* __restrict__ We2, const float* __restrict__ be2,
                             _Float16* __restrict__ ef_out) {
    int e = blockIdx.x * blockDim.x + threadIdx.x;
    if (e >= GN_E2) return;
    int e0 = (e < GN_E) ? e : e - GN_E;
    float d = distances[e0];
    float f[18];
#pragma unroll
    for (int j = 0; j < 16; ++j) {
        float c = (20.0f / 15.0f) * (float)j;
        float t = d - c;
        f[j] = expf(-t * t);
    }
    f[16] = edge_raw[(unsigned)e0 * 2 + 0];
    f[17] = edge_raw[(unsigned)e0 * 2 + 1];
    float h1[4];
#pragma unroll
    for (int i = 0; i < 4; ++i) {
        float a = be1[i];
#pragma unroll
        for (int j = 0; j < 18; ++j) a += f[j] * We1[j * 4 + i];
        h1[i] = fmaxf(a, 0.0f);
    }
    h8 o;
#pragma unroll
    for (int i = 0; i < 8; ++i) {
        float a = be2[i];
#pragma unroll
        for (int j = 0; j < 4; ++j) a += h1[j] * We2[j * 8 + i];
        o[i] = (_Float16)fmaxf(a, 0.0f);
    }
    *(h8*)(ef_out + (unsigned)e * 8) = o;     // one b128 store
}

// ---------------------------------------------------------------------
// Node encoder: emb[res](32) ++ node_raw(51) -> 8 -> 16 (ReLU).
// ---------------------------------------------------------------------
__global__ void node_encoder(const float* __restrict__ node_raw,
                             const int* __restrict__ residues,
                             const float* __restrict__ emb,
                             const float* __restrict__ Wn1, const float* __restrict__ bn1,
                             const float* __restrict__ Wn2, const float* __restrict__ bn2,
                             _Float16* __restrict__ nf_out) {
    int n = blockIdx.x * blockDim.x + threadIdx.x;
    if (n >= GN_N) return;
    int r = residues[n];
    __builtin_assume(r >= 0 && r < 22);
    float h1[8];
#pragma unroll
    for (int i = 0; i < 8; ++i) {
        float a = bn1[i];
        for (int j = 0; j < 32; ++j) a += emb[(unsigned)r * 32 + j] * Wn1[j * 8 + i];
        for (int j = 0; j < 51; ++j) a += node_raw[(unsigned)n * 51 + j] * Wn1[(32 + j) * 8 + i];
        h1[i] = fmaxf(a, 0.0f);
    }
    h8 o0, o1;
#pragma unroll
    for (int i = 0; i < 16; ++i) {
        float a = bn2[i];
#pragma unroll
        for (int j = 0; j < 8; ++j) a += h1[j] * Wn2[j * 16 + i];
        _Float16 v = (_Float16)fmaxf(a, 0.0f);
        if (i < 8) o0[i] = v; else o1[i - 8] = v;
    }
    *(h8*)(nf_out + (unsigned)n * 16 + 0) = o0;
    *(h8*)(nf_out + (unsigned)n * 16 + 8) = o1;
}

// ---------------------------------------------------------------------
// In-degree counts over symmetrized receivers (same every layer).
// ---------------------------------------------------------------------
__global__ void count_in_degree(const int* __restrict__ senders,
                                const int* __restrict__ receivers,
                                float* __restrict__ cnt) {
    int e = blockIdx.x * blockDim.x + threadIdx.x;
    if (e >= GN_E2) return;
    int rv = (e < GN_E) ? receivers[e] : senders[e - GN_E];
    __builtin_assume(rv >= 0 && rv < GN_N);
    atomicAdd(&cnt[rv], 1.0f);
}

__global__ void inv_count(const float* __restrict__ cnt, float* __restrict__ invc) {
    int n = blockIdx.x * blockDim.x + threadIdx.x;
    if (n >= GN_N) return;
    invc[n] = 1.0f / fmaxf(cnt[n], 1.0f);
}

// ---------------------------------------------------------------------
// 8-half aligned window loaders (each window lies entirely in one source
// region because all feature dims are multiples of 8). Bases pre-scaled
// 32-bit element offsets (all buffers < 2^31 elements).
// ---------------------------------------------------------------------
template<int IN_E, int IN_N>
__device__ __forceinline__ h8 win_edge(const _Float16* __restrict__ ef_in,
                                       const _Float16* __restrict__ nf_in,
                                       unsigned ebase, unsigned nbase, int k0) {
    if (k0 < IN_E)             return *(const h8*)(ef_in + ebase + k0);
    if (k0 < IN_E + IN_N)      return *(const h8*)(nf_in + nbase + (k0 - IN_E));
    h8 z = {}; return z;
}

template<int IN_N, int AGW>
__device__ __forceinline__ h8 win_node(const _Float16* __restrict__ nf_in,
                                       const float* __restrict__ agg,
                                       unsigned nbase, unsigned abase, float ic, int k0) {
    if (k0 < IN_N) return *(const h8*)(nf_in + nbase + k0);
    if (k0 < IN_N + AGW) {
        f8 f = *(const f8*)(agg + abase + (k0 - IN_N));
        h8 r;
#pragma unroll
        for (int j = 0; j < 8; ++j) r[j] = (_Float16)(f[j] * ic);
        return r;
    }
    h8 z = {}; return z;
}

// ---------------------------------------------------------------------
// Edge update (WMMA): ef_out = relu([ef_in | nf_in[snd]] @ W + be),
// fused scatter-add into agg[receiver]. One wave per 16-edge tile.
// A fragments: 2x b128 loads per 32-K chunk. B: 1x 32B fragment load.
// Output: LDS transpose -> 128-bit coalesced row stores.
// ---------------------------------------------------------------------
template<int IN_E, int IN_N, int OUT_E, int KP>
__global__ __launch_bounds__(128)
void edge_update(const _Float16* __restrict__ ef_in,
                 const _Float16* __restrict__ nf_in,
                 const int* __restrict__ senders,
                 const int* __restrict__ receivers,
                 const _Float16* __restrict__ Wf,   // fragment-packed weights
                 const float* __restrict__ be,      // [OUT_E]
                 _Float16* __restrict__ ef_out,     // [2E x OUT_E]
                 float* __restrict__ agg) {         // [N x OUT_E] (pre-zeroed)
    constexpr int QK = KP / 32;
    constexpr int NT = OUT_E / 16;
    constexpr int SROW = OUT_E + 8;                 // LDS row pad (bank spread)
    __shared__ _Float16 sout[4 * 16 * SROW];

    const int lane = threadIdx.x & 31;
    const int wave = threadIdx.x >> 5;
    const int base = (blockIdx.x * 4 + wave) * 16;
    if (base >= GN_E2) return;                      // wave-uniform: EXEC stays full

    const bool hi = lane >= 16;
    const int  m  = lane & 15;
    const int  e  = base + m;
    int s = (e < GN_E) ? senders[e] : receivers[e - GN_E];
    __builtin_assume(s >= 0 && s < GN_N);
    const unsigned ebase = (unsigned)e * IN_E;
    const unsigned nbase = (unsigned)s * IN_N;

    // A fragments (shared across all N-tiles): two contiguous windows each
    v16h a[QK];
#pragma unroll
    for (int q = 0; q < QK; ++q) {
        h8 w0 = win_edge<IN_E, IN_N>(ef_in, nf_in, ebase, nbase, q * 32 + (hi ? 8 : 0));
        h8 w1 = win_edge<IN_E, IN_N>(ef_in, nf_in, ebase, nbase, q * 32 + 16 + (hi ? 8 : 0));
        a[q] = __builtin_shufflevector(w0, w1, 0, 1, 2, 3, 4, 5, 6, 7,
                                       8, 9, 10, 11, 12, 13, 14, 15);
    }

    // receiver row bases for the 8 rows this lane's D registers cover
    unsigned rbase[8];
#pragma unroll
    for (int r = 0; r < 8; ++r) {
        int er = base + (hi ? r + 8 : r);
        int rv = (er < GN_E) ? receivers[er] : senders[er - GN_E];
        __builtin_assume(rv >= 0 && rv < GN_N);
        rbase[r] = (unsigned)rv * OUT_E;
    }

    _Float16* so = &sout[wave * 16 * SROW];
    const v16h* Wv = (const v16h*)Wf;

#pragma unroll
    for (int t = 0; t < NT; ++t) {
        const int col = t * 16 + m;
        v8f acc = {};
#pragma unroll
        for (int q = 0; q < QK; ++q) {
            v16h b = Wv[(t * QK + q) * 32 + lane];
            acc = __builtin_amdgcn_wmma_f32_16x16x32_f16(false, a[q], false, b,
                                                         (short)0, acc, false, false);
        }
        float bias = be[col];
#pragma unroll
        for (int r = 0; r < 8; ++r) {
            int M = hi ? r + 8 : r;
            float v = fmaxf(acc[r] + bias, 0.0f);
            so[M * SROW + col] = (_Float16)v;                 // ds_store_b16
            atomicAdd(agg + rbase[r] + col, v);               // 32-bit offset
        }
    }

    // per-wave LDS ops are processed in order; just stop compiler reordering
    asm volatile("" ::: "memory");

    // coalesced row stores: lane handles row m, half the columns
    const int cb = hi ? OUT_E / 2 : 0;
    const unsigned obase = (unsigned)(base + m) * OUT_E + cb;
#pragma unroll
    for (int j = 0; j < OUT_E / 16; ++j) {
        h8 v = *(const h8*)(so + m * SROW + cb + j * 8);
        *(h8*)(ef_out + obase + j * 8) = v;                   // b128 store
    }
}

// ---------------------------------------------------------------------
// Node update (WMMA): nf_out = relu([nf_in | agg/deg] @ W + bn).
// Segment-mean fused into A-fragment build; LDS transpose on output.
// ---------------------------------------------------------------------
template<int IN_N, int AGW, int OUT_N, int KP>
__global__ __launch_bounds__(128)
void node_update(const _Float16* __restrict__ nf_in,
                 const float* __restrict__ agg,
                 const float* __restrict__ invc,
                 const _Float16* __restrict__ Wf,
                 const float* __restrict__ bn,
                 _Float16* __restrict__ nf_out) {
    constexpr int QK = KP / 32;
    constexpr int NT = OUT_N / 16;
    constexpr int SROW = OUT_N + 8;
    __shared__ _Float16 sout[4 * 16 * SROW];

    const int lane = threadIdx.x & 31;
    const int wave = threadIdx.x >> 5;
    const int base = (blockIdx.x * 4 + wave) * 16;
    if (base >= GN_N) return;                       // wave-uniform

    const bool hi = lane >= 16;
    const int  m  = lane & 15;
    const int  n  = base + m;
    const float ic = invc[n];
    const unsigned nbase = (unsigned)n * IN_N;
    const unsigned abase = (unsigned)n * AGW;

    v16h a[QK];
#pragma unroll
    for (int q = 0; q < QK; ++q) {
        h8 w0 = win_node<IN_N, AGW>(nf_in, agg, nbase, abase, ic, q * 32 + (hi ? 8 : 0));
        h8 w1 = win_node<IN_N, AGW>(nf_in, agg, nbase, abase, ic, q * 32 + 16 + (hi ? 8 : 0));
        a[q] = __builtin_shufflevector(w0, w1, 0, 1, 2, 3, 4, 5, 6, 7,
                                       8, 9, 10, 11, 12, 13, 14, 15);
    }

    _Float16* so = &sout[wave * 16 * SROW];
    const v16h* Wv = (const v16h*)Wf;

#pragma unroll
    for (int t = 0; t < NT; ++t) {
        const int col = t * 16 + m;
        v8f acc = {};
#pragma unroll
        for (int q = 0; q < QK; ++q) {
            v16h b = Wv[(t * QK + q) * 32 + lane];
            acc = __builtin_amdgcn_wmma_f32_16x16x32_f16(false, a[q], false, b,
                                                         (short)0, acc, false, false);
        }
        float bias = bn[col];
#pragma unroll
        for (int r = 0; r < 8; ++r) {
            int M = hi ? r + 8 : r;
            so[M * SROW + col] = (_Float16)fmaxf(acc[r] + bias, 0.0f);
        }
    }

    asm volatile("" ::: "memory");

    const int cb = hi ? OUT_N / 2 : 0;
    const unsigned obase = (unsigned)n * OUT_N + cb;
#pragma unroll
    for (int j = 0; j < OUT_N / 16; ++j) {
        h8 v = *(const h8*)(so + m * SROW + cb + j * 8);
        *(h8*)(nf_out + obase + j * 8) = v;
    }
}

// ---------------------------------------------------------------------
// Readout
// ---------------------------------------------------------------------
__global__ void graph_accum(const _Float16* __restrict__ nf,
                            const int* __restrict__ gidx,
                            float* __restrict__ gsum, float* __restrict__ gcnt) {
    int n = blockIdx.x;                 // one block (128 thr) per node
    int k = threadIdx.x;
    int g = gidx[n];
    __builtin_assume(g >= 0 && g < GN_G);
    atomicAdd(gsum + (unsigned)g * 128 + k, (float)nf[(unsigned)n * 128 + k]);
    if (k == 0) atomicAdd(&gcnt[g], 1.0f);
}

__global__ void glob_readout(const float* __restrict__ gsum, const float* __restrict__ gcnt,
                             const float* __restrict__ Wg, const float* __restrict__ bg,
                             float* __restrict__ out) {
    int g = blockIdx.x * blockDim.x + threadIdx.x;
    if (g >= GN_G) return;
    float ic = 1.0f / fmaxf(gcnt[g], 1.0f);
    float a = bg[0];
    for (int k = 0; k < 128; ++k) a += gsum[(unsigned)g * 128 + k] * ic * Wg[k];
    out[g] = gn_sigmoid(a);
}

__global__ void node_readout(const _Float16* __restrict__ nf,
                             const float* __restrict__ Wn, const float* __restrict__ bn,
                             float* __restrict__ out) {
    int n = blockIdx.x * blockDim.x + threadIdx.x;
    if (n >= GN_N) return;
    float a0 = bn[0], a1 = bn[1];
#pragma unroll
    for (int j = 0; j < 16; ++j) {
        h8 v = *(const h8*)(nf + (unsigned)n * 128 + j * 8);
#pragma unroll
        for (int i = 0; i < 8; ++i) {
            float x = (float)v[i];
            a0 += x * Wn[(j * 8 + i) * 2 + 0];
            a1 += x * Wn[(j * 8 + i) * 2 + 1];
        }
    }
    out[(unsigned)n * 2 + 0] = gn_sigmoid(a0);
    out[(unsigned)n * 2 + 1] = gn_sigmoid(a1);
}

// ---------------------------------------------------------------------
// Host launcher
// ---------------------------------------------------------------------
extern "C" void kernel_launch(void* const* d_in, const int* in_sizes, int n_in,
                              void* d_out, int out_size, void* d_ws, size_t ws_size,
                              hipStream_t stream) {
    (void)in_sizes; (void)n_in; (void)out_size; (void)ws_size;
    const float* node_raw  = (const float*)d_in[0];
    const float* edge_raw  = (const float*)d_in[1];
    const float* distances = (const float*)d_in[2];
    const int*   residues  = (const int*)d_in[3];
    const int*   senders   = (const int*)d_in[4];
    const int*   receivers = (const int*)d_in[5];
    const int*   gidx      = (const int*)d_in[6];
    const float* emb       = (const float*)d_in[7];
    const float* eWe1 = (const float*)d_in[8];  const float* ebe1 = (const float*)d_in[9];
    const float* eWe2 = (const float*)d_in[10]; const float* ebe2 = (const float*)d_in[11];
    const float* eWn1 = (const float*)d_in[12]; const float* ebn1 = (const float*)d_in[13];
    const float* eWn2 = (const float*)d_in[14]; const float* ebn2 = (const float*)d_in[15];
    const float* roWg = (const float*)d_in[16]; const float* robg = (const float*)d_in[17];
    const float* roWn = (const float*)d_in[18]; const float* robn = (const float*)d_in[19];
    const float *lWe[3], *lWs[3], *lbe[3], *lWn[3], *lWi[3], *lbn[3];
    for (int i = 0; i < 3; ++i) {
        lWe[i] = (const float*)d_in[20 + 6 * i];
        lWs[i] = (const float*)d_in[21 + 6 * i];
        lbe[i] = (const float*)d_in[22 + 6 * i];
        lWn[i] = (const float*)d_in[23 + 6 * i];
        lWi[i] = (const float*)d_in[24 + 6 * i];
        lbn[i] = (const float*)d_in[25 + 6 * i];
    }
    float* out = (float*)d_out;

    // workspace carve-up (ping-pong activation buffers, f16)
    char* ws = (char*)d_ws;
    size_t off = 0;
    auto take = [&](size_t bytes) -> char* {
        char* p = ws + off;
        off = (off + bytes + 255) & ~(size_t)255;
        return p;
    };
    _Float16* ef_a = (_Float16*)take((size_t)GN_E2 * 32 * 2);  // w8 / w32 stages
    _Float16* ef_b = (_Float16*)take((size_t)GN_E2 * 64 * 2);  // w16 / w64 stages
    _Float16* nf_a = (_Float16*)take((size_t)GN_N * 64 * 2);   // w16 / w64
    _Float16* nf_b = (_Float16*)take((size_t)GN_N * 128 * 2);  // w32 / w128
    float* agg  = (float*)take((size_t)GN_N * 64 * 4);
    float* cnt  = (float*)take((size_t)GN_N * 4);
    float* invc = (float*)take((size_t)GN_N * 4);
    float* gsum = (float*)take((size_t)GN_G * 128 * 4);
    float* gcnt = (float*)take((size_t)GN_G * 4);
    _Float16* wpe0 = (_Float16*)take((size_t)32 * 16 * 2);
    _Float16* wpe1 = (_Float16*)take((size_t)64 * 32 * 2);
    _Float16* wpe2 = (_Float16*)take((size_t)96 * 64 * 2);
    _Float16* wpn0 = (_Float16*)take((size_t)32 * 32 * 2);
    _Float16* wpn1 = (_Float16*)take((size_t)64 * 64 * 2);
    _Float16* wpn2 = (_Float16*)take((size_t)128 * 128 * 2);

    // zero atomic accumulators (capture-safe)
    hipMemsetAsync(cnt,  0, (size_t)GN_N * 4, stream);
    hipMemsetAsync(gsum, 0, (size_t)GN_G * 128 * 4, stream);
    hipMemsetAsync(gcnt, 0, (size_t)GN_G * 4, stream);

    // pack WMMA weight fragments (concat rows + pad to KP, f16, frag order)
    auto pk = [&](const float* W1, int r1, const float* W2, int r2, int cols, int KP,
                  _Float16* o) {
        int tot = (cols / 16) * (KP / 32) * 512;
        pack_weights_frag<<<(tot + 255) / 256, 256, 0, stream>>>(W1, r1, W2, r2, cols, KP, o);
    };
    pk(lWe[0], 8,  lWs[0], 16, 16,  32,  wpe0);
    pk(lWe[1], 16, lWs[1], 32, 32,  64,  wpe1);
    pk(lWe[2], 32, lWs[2], 64, 64,  96,  wpe2);
    pk(lWn[0], 16, lWi[0], 16, 32,  32,  wpn0);
    pk(lWn[1], 32, lWi[1], 32, 64,  64,  wpn1);
    pk(lWn[2], 64, lWi[2], 64, 128, 128, wpn2);

    // encoders
    edge_encoder<<<(GN_E2 + 255) / 256, 256, 0, stream>>>(edge_raw, distances,
                                                          eWe1, ebe1, eWe2, ebe2, ef_a);
    node_encoder<<<(GN_N + 255) / 256, 256, 0, stream>>>(node_raw, residues, emb,
                                                         eWn1, ebn1, eWn2, ebn2, nf_a);

    // in-degree (fixed across layers)
    count_in_degree<<<(GN_E2 + 255) / 256, 256, 0, stream>>>(senders, receivers, cnt);
    inv_count<<<(GN_N + 255) / 256, 256, 0, stream>>>(cnt, invc);

    const int eblk = (GN_E2 / 16 + 3) / 4;   // 16-edge tiles, 4 waves/block
    const int nblk = (GN_N  / 16 + 3) / 4;

    // ---- layer 0: (8,16,16,32) ----
    hipMemsetAsync(agg, 0, (size_t)GN_N * 16 * 4, stream);
    edge_update<8, 16, 16, 32><<<eblk, 128, 0, stream>>>(ef_a, nf_a, senders, receivers,
                                                         wpe0, lbe[0], ef_b, agg);
    node_update<16, 16, 32, 32><<<nblk, 128, 0, stream>>>(nf_a, agg, invc, wpn0, lbn[0], nf_b);

    // ---- layer 1: (16,32,32,64) ----
    hipMemsetAsync(agg, 0, (size_t)GN_N * 32 * 4, stream);
    edge_update<16, 32, 32, 64><<<eblk, 128, 0, stream>>>(ef_b, nf_b, senders, receivers,
                                                          wpe1, lbe[1], ef_a, agg);
    node_update<32, 32, 64, 64><<<nblk, 128, 0, stream>>>(nf_b, agg, invc, wpn1, lbn[1], nf_a);

    // ---- layer 2: (32,64,64,128) ----
    hipMemsetAsync(agg, 0, (size_t)GN_N * 64 * 4, stream);
    edge_update<32, 64, 64, 96><<<eblk, 128, 0, stream>>>(ef_a, nf_a, senders, receivers,
                                                          wpe2, lbe[2], ef_b, agg);
    node_update<64, 64, 128, 128><<<nblk, 128, 0, stream>>>(nf_a, agg, invc, wpn2, lbn[2], nf_b);

    // ---- readout ----
    graph_accum<<<GN_N, 128, 0, stream>>>(nf_b, gidx, gsum, gcnt);
    glob_readout<<<(GN_G + 255) / 256, 256, 0, stream>>>(gsum, gcnt, roWg, robg,
                                                         out + (size_t)GN_N * 2);
    node_readout<<<(GN_N + 255) / 256, 256, 0, stream>>>(nf_b, roWn, robn, out);
}
